// Mamba2Mixer_22393959481413
// MI455X (gfx1250) — matrix operand order repaired and checked by reading, not compiled, not run on hardware
//
#include <hip/hip_runtime.h>
#include <hip/hip_bf16.h>

#define BB 2
#define LL 4096
#define HH 1024
#define INTER 2048
#define NSTATE 128
#define HEADS 32
#define PP 64
#define KK 4
#define CHUNK 256
#define NC (LL / CHUNK)                 // 16
#define CONV_DIM (INTER + 2 * NSTATE)   // 2304
#define PROJ (INTER + CONV_DIM + HEADS) // 4384
#define PROJP 4480                      // padded to multiple of 128
#define BL (BB * LL)                    // 8192

typedef __attribute__((ext_vector_type(16))) _Float16 v16h;
typedef __attribute__((ext_vector_type(8)))  _Float16 h8;
typedef __attribute__((ext_vector_type(8)))  float    v8f;

__device__ __forceinline__ v8f wmma16(v16h a, v16h b, v8f c) {
  return __builtin_amdgcn_wmma_f32_16x16x32_f16(false, a, false, b, (short)0, c,
                                                false, false);
}

__device__ __forceinline__ v8f vzero() {
  v8f z = {0.f, 0.f, 0.f, 0.f, 0.f, 0.f, 0.f, 0.f};
  return z;
}

// Async DMA: global memory -> LDS, 16 bytes per lane, tracked by ASYNCcnt.
__device__ __forceinline__ void async_b128(unsigned lds_addr, const void* gaddr) {
  asm volatile("global_load_async_to_lds_b128 %0, %1, off"
               :: "v"(lds_addr), "v"((unsigned long long)(uintptr_t)gaddr)
               : "memory");
}

// A tile (16x32, f16): A[m][k] from row-major src (ld elements per row).
// Layout: lanes 0-15 -> M=lane, K lo-half {0..7,16..23}; lanes 16-31 K {8..15,24..31}.
__device__ __forceinline__ v16h load_a_rm(const _Float16* src, int ld) {
  const int lane = threadIdx.x & 31;
  const int m = lane & 15;
  const int kb = (lane < 16) ? 0 : 8;
  const _Float16* p = src + (size_t)m * ld;
  v16h a;
#pragma unroll
  for (int j = 0; j < 16; ++j) {
    int k = (j < 8) ? (kb + j) : (kb + 8 + j);
    a[j] = p[k];
  }
  return a;
}

// A tile where storage is [k][m] row-major (transpose load).
__device__ __forceinline__ v16h load_a_tr(const _Float16* src, int ld) {
  const int lane = threadIdx.x & 31;
  const int m = lane & 15;
  const int kb = (lane < 16) ? 0 : 8;
  v16h a;
#pragma unroll
  for (int j = 0; j < 16; ++j) {
    int k = (j < 8) ? (kb + j) : (kb + 8 + j);
    a[j] = src[(size_t)k * ld + m];
  }
  return a;
}

// B tile (32x16, f16): B[k][n] from row-major [k][n] storage.
// Layout: lanes 0-15 -> N=lane, K {0..7,16..23}; lanes 16-31 K {8..15,24..31}.
__device__ __forceinline__ v16h load_b_rm(const _Float16* src, int ld) {
  const int lane = threadIdx.x & 31;
  const int n = lane & 15;
  const int kb = (lane < 16) ? 0 : 8;
  v16h b;
#pragma unroll
  for (int j = 0; j < 16; ++j) b[j] = src[(size_t)((j < 8) ? (kb + j) : (kb + 8 + j)) * ld + n];
  return b;
}

// B tile where storage is [n][k] row-major (e.g. weight matrices W[N][K]).
__device__ __forceinline__ v16h load_b_tr(const _Float16* src, int ld) {
  const int lane = threadIdx.x & 31;
  const int n = lane & 15;
  const int kb = (lane < 16) ? 0 : 8;
  const _Float16* p = src + (size_t)n * ld;
  v16h b;
#pragma unroll
  for (int j = 0; j < 16; ++j) b[j] = p[(j < 8) ? (kb + j) : (kb + 8 + j)];
  return b;
}

// D/C f32 16x16: lanes 0-15 N=lane M=r; lanes 16-31 N=lane-16 M=r+8.
__device__ __forceinline__ void store_d_f32(float* dst, int ld, v8f d) {
  const int lane = threadIdx.x & 31;
  const int n = lane & 15;
  const int mb = (lane < 16) ? 0 : 8;
#pragma unroll
  for (int r = 0; r < 8; ++r) dst[(size_t)(mb + r) * ld + n] = d[r];
}

// ---------------- converts ----------------
__global__ void cvt_f16_kernel(const float* __restrict__ src,
                               _Float16* __restrict__ dst, int n) {
  int i = blockIdx.x * 256 + threadIdx.x;
  if (i < n) dst[i] = (_Float16)src[i];
}

__global__ void cvt_win_kernel(const float* __restrict__ w,
                               _Float16* __restrict__ dst) {
  int i = blockIdx.x * 256 + threadIdx.x;   // over PROJP*HH
  int row = i >> 10, col = i & 1023;
  if (row < PROJP)
    dst[i] = (row < PROJ) ? (_Float16)w[(size_t)row * HH + col] : (_Float16)0.f;
}

// ------- 128x128 WMMA GEMM, double-buffered LDS fed by async-to-LDS DMA ------
// C[M][ldc] = A[M][K] * W[N][K]^T
__global__ void __launch_bounds__(256)
gemm_kernel(const _Float16* __restrict__ A, const _Float16* __restrict__ W,
            float* __restrict__ C, int K, int ldc) {
  __shared__ _Float16 sA[2][128 * 32];   // 8 KB per buffer
  __shared__ _Float16 sB[2][128 * 32];
  const int m0 = blockIdx.y * 128;
  const int n0 = blockIdx.x * 128;
  const int t = threadIdx.x;
  const int w = t >> 5;
  const int row = t >> 1;            // 0..127
  const int kk = (t & 1) * 16;       // 0 or 16 (halfs)

  const _Float16* gA = A + (size_t)(m0 + row) * K + kk;
  const _Float16* gB = W + (size_t)(n0 + row) * K + kk;
  const unsigned la = (unsigned)(uintptr_t)(void*)(&sA[0][row * 32 + kk]);
  const unsigned lb = (unsigned)(uintptr_t)(void*)(&sB[0][row * 32 + kk]);
  const unsigned bufB = (unsigned)(128 * 32 * sizeof(_Float16));  // 8192 bytes

  v8f acc[8];
#pragma unroll
  for (int i = 0; i < 8; ++i) acc[i] = vzero();

  // issue stage copies (4 async ops per wave per stage)
  auto issue = [&](int k0, int buf) {
    const unsigned o = buf ? bufB : 0u;
    async_b128(la + o,      gA + k0);
    async_b128(la + o + 16, gA + k0 + 8);
    async_b128(lb + o,      gB + k0);
    async_b128(lb + o + 16, gB + k0 + 8);
  };

  issue(0, 0);
  const int nstages = K / 32;
  for (int s = 0; s < nstages; ++s) {
    const int cur = s & 1;
    if (s + 1 < nstages) {
      issue((s + 1) * 32, cur ^ 1);            // prefetch next stage via DMA
      asm volatile("s_wait_asynccnt 0x4" ::: "memory");  // cur stage landed
    } else {
      asm volatile("s_wait_asynccnt 0x0" ::: "memory");
    }
    __syncthreads();                            // all waves' DMA visible
    v16h b = load_b_tr(&sB[cur][(w * 16) * 32], 32);
#pragma unroll
    for (int mi = 0; mi < 8; ++mi) {
      v16h a = load_a_rm(&sA[cur][(mi * 16) * 32], 32);
      acc[mi] = wmma16(a, b, acc[mi]);
    }
    __syncthreads();                            // done reading cur buffer
  }
#pragma unroll
  for (int mi = 0; mi < 8; ++mi)
    store_d_f32(C + (size_t)(m0 + mi * 16) * ldc + n0 + w * 16, ldc, acc[mi]);
}

// ---------------- conv1d + SiLU + splits + dt(softplus) ----------------
__global__ void __launch_bounds__(256)
postproj_kernel(const float* __restrict__ proj, const float* __restrict__ conv_w,
                const float* __restrict__ conv_b, const float* __restrict__ dt_bias,
                float* __restrict__ xf32, _Float16* __restrict__ xdt16,
                _Float16* __restrict__ b16, _Float16* __restrict__ c16,
                float* __restrict__ dtb) {
  const int bl = blockIdx.x;
  const int b = bl / LL, l = bl % LL;
  const int t = threadIdx.x;
  __shared__ float s_dt[HEADS];
  const float* prow = proj + (size_t)bl * PROJP;
  if (t < HEADS) {
    float x = prow[INTER + CONV_DIM + t] + dt_bias[t];
    float dt = (x > 20.f) ? x : log1pf(__expf(x));
    s_dt[t] = dt;
    dtb[((size_t)b * HEADS + t) * LL + l] = dt;
  }
  __syncthreads();
#pragma unroll
  for (int i = 0; i < CONV_DIM / 256; ++i) {   // 9 iterations
    int c = t + i * 256;
    float acc = conv_b[c];
    const float* wr = conv_w + c * KK;
#pragma unroll
    for (int tap = 0; tap < KK; ++tap) {
      int ll2 = l - (KK - 1) + tap;
      if (ll2 >= 0)
        acc += wr[tap] * proj[((size_t)b * LL + ll2) * PROJP + INTER + c];
    }
    float s = acc / (1.f + __expf(-acc));  // SiLU
    if (c < INTER) {
      int h = c >> 6;
      xf32[(size_t)bl * INTER + c] = s;
      xdt16[(size_t)bl * INTER + c] = (_Float16)(s * s_dt[h]);
    } else if (c < INTER + NSTATE) {
      b16[(size_t)bl * NSTATE + (c - INTER)] = (_Float16)s;
    } else {
      c16[(size_t)bl * NSTATE + (c - INTER - NSTATE)] = (_Float16)s;
    }
  }
}

// ---------------- per-(b,h,chunk) cumsum of dt*A; decay-weighted X ----------
__global__ void __launch_bounds__(256)
cumsum_kernel(const float* __restrict__ dtb, const float* __restrict__ A_log,
              float* __restrict__ acs, const _Float16* __restrict__ xdt16,
              _Float16* __restrict__ xdd16) {
  const int blk = blockIdx.x;           // (b*HEADS+h)*NC + c
  const int c = blk % NC;
  const int bh = blk / NC;
  const int h = bh % HEADS, b = bh / HEADS;
  const int s = threadIdx.x;
  const int l = c * CHUNK + s;
  __shared__ float sc[CHUNK];
  float Ah = -__expf(A_log[h]);
  sc[s] = dtb[((size_t)b * HEADS + h) * LL + l] * Ah;
  __syncthreads();
  for (int off = 1; off < CHUNK; off <<= 1) {
    float x = (s >= off) ? sc[s - off] : 0.f;
    __syncthreads();
    sc[s] += x;
    __syncthreads();
  }
  float a = sc[s];
  float tot = sc[CHUNK - 1];
  acs[(size_t)bh * LL + l] = a;
  _Float16 hd = (_Float16)__expf(tot - a);
  const _Float16* src = xdt16 + ((size_t)b * LL + l) * INTER + h * PP;
  _Float16* dst = xdd16 + ((size_t)b * LL + l) * INTER + h * PP;
#pragma unroll
  for (int p = 0; p < PP; ++p) dst[p] = src[p] * hd;
}

// ---------------- states[b,c,h,p,n] = sum_s Xdd[s,p] * B[s,n] ----------------
__global__ void __launch_bounds__(256)
states_kernel(const _Float16* __restrict__ xdd16, const _Float16* __restrict__ b16,
              float* __restrict__ states) {
  const int blk = blockIdx.x;           // (b*NC+c)*HEADS + h
  const int h = blk % HEADS;
  const int bc = blk / HEADS;
  const int w = threadIdx.x >> 5;       // n-tile 0..7
  const _Float16* Xd = xdd16 + (size_t)bc * CHUNK * INTER + h * PP;  // [s][p]
  const _Float16* Bc = b16 + (size_t)bc * CHUNK * NSTATE;            // [s][n]
  v8f acc[4];
#pragma unroll
  for (int i = 0; i < 4; ++i) acc[i] = vzero();
  for (int k0 = 0; k0 < CHUNK; k0 += 32) {
    v16h bt = load_b_rm(Bc + (size_t)k0 * NSTATE + w * 16, NSTATE);
#pragma unroll
    for (int mi = 0; mi < 4; ++mi) {
      v16h a = load_a_tr(Xd + (size_t)k0 * INTER + mi * 16, INTER);
      acc[mi] = wmma16(a, bt, acc[mi]);
    }
  }
  float* out = states + (size_t)blk * (PP * NSTATE) + w * 16;
#pragma unroll
  for (int mi = 0; mi < 4; ++mi)
    store_d_f32(out + (size_t)(mi * 16) * NSTATE, NSTATE, acc[mi]);
}

// ---------------- inter-chunk recurrence (16 steps, sequential) --------------
__global__ void __launch_bounds__(256)
scan_kernel(const float* __restrict__ states, const float* __restrict__ acs,
            _Float16* __restrict__ prev16) {
  const int bh = blockIdx.x;            // b*HEADS + h
  const int b = bh / HEADS, h = bh % HEADS;
  const int t = threadIdx.x;
  float carry[32];
#pragma unroll
  for (int i = 0; i < 32; ++i) carry[i] = 0.f;
  for (int c = 0; c < NC; ++c) {
    size_t base = (((size_t)(b * NC + c) * HEADS) + h) * (PP * NSTATE);
    float dec = __expf(acs[(size_t)bh * LL + c * CHUNK + CHUNK - 1]);
#pragma unroll
    for (int i = 0; i < 32; ++i) {
      int idx = t + i * 256;
      prev16[base + idx] = (_Float16)carry[i];
      carry[i] = carry[i] * dec + states[base + idx];
    }
  }
}

// ---------------- CB[l,s] = sum_n C[l,n] B[s,n]  (per b,c; shared by heads) --
__global__ void __launch_bounds__(256)
cb_kernel(const _Float16* __restrict__ c16, const _Float16* __restrict__ b16,
          _Float16* __restrict__ cbg) {
  const int bc = blockIdx.x;            // b*NC + c
  const int w = threadIdx.x >> 5;
  const int lane = threadIdx.x & 31;
  const int nn = lane & 15;
  const int mb = (lane < 16) ? 0 : 8;
  const _Float16* Cp = c16 + (size_t)bc * CHUNK * NSTATE;
  const _Float16* Bp = b16 + (size_t)bc * CHUNK * NSTATE;
  _Float16* out = cbg + (size_t)bc * CHUNK * CHUNK;
  for (int mt = w * 2; mt < w * 2 + 2; ++mt) {
    for (int st = 0; st < 16; ++st) {
      v8f acc = vzero();
#pragma unroll
      for (int k0 = 0; k0 < NSTATE; k0 += 32) {
        v16h a = load_a_rm(Cp + (size_t)(mt * 16) * NSTATE + k0, NSTATE);
        v16h bt = load_b_tr(Bp + (size_t)(st * 16) * NSTATE + k0, NSTATE);
        acc = wmma16(a, bt, acc);
      }
#pragma unroll
      for (int r = 0; r < 8; ++r)
        out[(size_t)(mt * 16 + mb + r) * CHUNK + st * 16 + nn] = (_Float16)acc[r];
    }
  }
}

// ------- Y = (CB o Lm) @ Xdt  +  (C o exp(Acs)) @ prev^T  +  D*x  ------------
__global__ void __launch_bounds__(256)
y_kernel(const _Float16* __restrict__ cbg, const _Float16* __restrict__ xdt16,
         const _Float16* __restrict__ c16, const _Float16* __restrict__ prev16,
         const float* __restrict__ acs, const float* __restrict__ xf32,
         const float* __restrict__ Dp, float* __restrict__ ybuf) {
  const int blk = blockIdx.x;           // (b*NC+c)*HEADS + h
  const int h = blk % HEADS;
  const int bc = blk / HEADS;
  const int b = bc / NC, c = bc % NC;
  const int t = threadIdx.x;
  const int w = t >> 5;
  const int lane = t & 31;
  const int mrow = lane & 15;
  const int kb8 = (lane < 16) ? 0 : 8;
  const int nn = lane & 15;
  const int mb = kb8;
  __shared__ float sAcs[CHUNK];
  sAcs[t] = acs[((size_t)(b * HEADS + h)) * LL + c * CHUNK + t];
  __syncthreads();

  const _Float16* CB = cbg + (size_t)bc * CHUNK * CHUNK;
  const _Float16* Xd = xdt16 + (size_t)bc * CHUNK * INTER + h * PP;
  const _Float16* Cp = c16 + (size_t)bc * CHUNK * NSTATE;
  const _Float16* Pv = prev16 + (size_t)blk * (PP * NSTATE);

  v8f acc[2][4];
#pragma unroll
  for (int mi = 0; mi < 2; ++mi)
#pragma unroll
    for (int nt = 0; nt < 4; ++nt) acc[mi][nt] = vzero();

#pragma unroll
  for (int mi = 0; mi < 2; ++mi) {
    const int mt = w * 2 + mi;
    const int l = mt * 16 + mrow;
    const float acs_l = sAcs[l];
    const _Float16* cbrow = CB + (size_t)l * CHUNK;
    // GEMM1: Yd, K = s over chunk
    for (int k0 = 0; k0 < CHUNK; k0 += 32) {
      v16h a;
#pragma unroll
      for (int j = 0; j < 16; ++j) {
        int s = k0 + ((j < 8) ? (kb8 + j) : (kb8 + 8 + j));
        float wgt = (s <= l) ? __expf(acs_l - sAcs[s]) : 0.f;
        a[j] = (_Float16)((float)cbrow[s] * wgt);
      }
#pragma unroll
      for (int nt = 0; nt < 4; ++nt) {
        v16h bt = load_b_rm(Xd + (size_t)k0 * INTER + nt * 16, INTER);
        acc[mi][nt] = wmma16(a, bt, acc[mi][nt]);
      }
    }
    // GEMM2: Yo, K = n over NSTATE
    const float scl = __expf(acs_l);
    const _Float16* crow = Cp + (size_t)l * NSTATE;
    for (int k0 = 0; k0 < NSTATE; k0 += 32) {
      v16h a;
#pragma unroll
      for (int j = 0; j < 16; ++j) {
        int k = k0 + ((j < 8) ? (kb8 + j) : (kb8 + 8 + j));
        a[j] = (_Float16)((float)crow[k] * scl);
      }
#pragma unroll
      for (int nt = 0; nt < 4; ++nt) {
        v16h bt = load_b_tr(Pv + (size_t)(nt * 16) * NSTATE + k0, NSTATE);
        acc[mi][nt] = wmma16(a, bt, acc[mi][nt]);
      }
    }
  }

  const float Dh = Dp[h];
#pragma unroll
  for (int mi = 0; mi < 2; ++mi) {
    const int mt = w * 2 + mi;
#pragma unroll
    for (int nt = 0; nt < 4; ++nt) {
#pragma unroll
      for (int r = 0; r < 8; ++r) {
        int lloc = mt * 16 + mb + r;
        size_t row = (size_t)b * LL + c * CHUNK + lloc;
        int col = h * PP + nt * 16 + nn;
        ybuf[row * INTER + col] = acc[mi][nt][r] + Dh * xf32[row * INTER + col];
      }
    }
  }
}

// ---------------- LayerNorm + SiLU gate -> f16 ----------------
__global__ void __launch_bounds__(256)
ln_gate_kernel(const float* __restrict__ ybuf, const float* __restrict__ proj,
               const float* __restrict__ ln_w, const float* __restrict__ ln_b,
               _Float16* __restrict__ g16) {
  const int bl = blockIdx.x;
  const int t = threadIdx.x;
  const float* y = ybuf + (size_t)bl * INTER;
  const float* g = proj + (size_t)bl * PROJP;
  __shared__ float red[256];
  float vals[8];
  float lsum = 0.f;
#pragma unroll
  for (int i = 0; i < 8; ++i) { vals[i] = y[t + i * 256]; lsum += vals[i]; }
  red[t] = lsum;
  __syncthreads();
  for (int off = 128; off > 0; off >>= 1) {
    if (t < off) red[t] += red[t + off];
    __syncthreads();
  }
  float mu = red[0] * (1.f / INTER);
  __syncthreads();
  float lvar = 0.f;
#pragma unroll
  for (int i = 0; i < 8; ++i) { float d = vals[i] - mu; lvar += d * d; }
  red[t] = lvar;
  __syncthreads();
  for (int off = 128; off > 0; off >>= 1) {
    if (t < off) red[t] += red[t + off];
    __syncthreads();
  }
  float inv = rsqrtf(red[0] * (1.f / INTER) + 1e-5f);
#pragma unroll
  for (int i = 0; i < 8; ++i) {
    int col = t + i * 256;
    float yn = (vals[i] - mu) * inv * ln_w[col] + ln_b[col];
    float gv = g[col];
    float sg = gv / (1.f + __expf(-gv));
    g16[(size_t)bl * INTER + col] = (_Float16)(yn * sg);
  }
}

// ---------------- launch ----------------
extern "C" void kernel_launch(void* const* d_in, const int* in_sizes, int n_in,
                              void* d_out, int out_size, void* d_ws, size_t ws_size,
                              hipStream_t stream) {
  const float* hs      = (const float*)d_in[0];
  const float* w_in    = (const float*)d_in[1];
  const float* conv_w  = (const float*)d_in[2];
  const float* conv_b  = (const float*)d_in[3];
  const float* dt_bias = (const float*)d_in[4];
  const float* A_log   = (const float*)d_in[5];
  const float* Dp      = (const float*)d_in[6];
  const float* ln_w    = (const float*)d_in[7];
  const float* ln_b    = (const float*)d_in[8];
  const float* w_out   = (const float*)d_in[9];
  float* out = (float*)d_out;

  char* ws = (char*)d_ws;
  size_t off = 0;
  auto carve = [&](size_t bytes) -> void* {
    void* p = ws + off;
    off = (off + bytes + 255) & ~(size_t)255;
    return p;
  };
  _Float16* hs16   = (_Float16*)carve((size_t)BL * HH * 2);
  _Float16* win16  = (_Float16*)carve((size_t)PROJP * HH * 2);
  _Float16* wo16   = (_Float16*)carve((size_t)HH * INTER * 2);
  float*    proj   = (float*)   carve((size_t)BL * PROJP * 4);
  float*    xf32   = (float*)   carve((size_t)BL * INTER * 4);
  _Float16* xdt16  = (_Float16*)carve((size_t)BL * INTER * 2);
  _Float16* xdd16  = (_Float16*)carve((size_t)BL * INTER * 2);
  _Float16* b16    = (_Float16*)carve((size_t)BL * NSTATE * 2);
  _Float16* c16    = (_Float16*)carve((size_t)BL * NSTATE * 2);
  float*    dtb    = (float*)   carve((size_t)BB * HEADS * LL * 4);
  float*    acs    = (float*)   carve((size_t)BB * HEADS * LL * 4);
  _Float16* cbg    = (_Float16*)carve((size_t)BB * NC * CHUNK * CHUNK * 2);
  float*    states = (float*)   carve((size_t)BB * NC * HEADS * PP * NSTATE * 4);
  _Float16* prev16 = (_Float16*)carve((size_t)BB * NC * HEADS * PP * NSTATE * 2);
  float*    ybuf   = (float*)   carve((size_t)BL * INTER * 4);
  _Float16* g16    = (_Float16*)carve((size_t)BL * INTER * 2);

  cvt_f16_kernel<<<(BL * HH) / 256, 256, 0, stream>>>(hs, hs16, BL * HH);
  cvt_win_kernel<<<(PROJP * HH) / 256, 256, 0, stream>>>(w_in, win16);
  cvt_f16_kernel<<<(HH * INTER) / 256, 256, 0, stream>>>(w_out, wo16, HH * INTER);

  dim3 g1(PROJP / 128, BL / 128);   // 35 x 64
  gemm_kernel<<<g1, 256, 0, stream>>>(hs16, win16, proj, HH, PROJP);

  postproj_kernel<<<BL, 256, 0, stream>>>(proj, conv_w, conv_b, dt_bias,
                                          xf32, xdt16, b16, c16, dtb);
  cumsum_kernel<<<BB * HEADS * NC, 256, 0, stream>>>(dtb, A_log, acs, xdt16, xdd16);
  cb_kernel<<<BB * NC, 256, 0, stream>>>(c16, b16, cbg);
  states_kernel<<<BB * NC * HEADS, 256, 0, stream>>>(xdd16, b16, states);
  scan_kernel<<<BB * HEADS, 256, 0, stream>>>(states, acs, prev16);
  y_kernel<<<BB * NC * HEADS, 256, 0, stream>>>(cbg, xdt16, c16, prev16, acs,
                                                xf32, Dp, ybuf);
  ln_gate_kernel<<<BL, 256, 0, stream>>>(ybuf, proj, ln_w, ln_b, g16);

  dim3 g2(HH / 128, BL / 128);      // 8 x 64
  gemm_kernel<<<g2, 256, 0, stream>>>(g16, wo16, out, INTER, HH);
}
